// LowRankStateMixing_14310831030384
// MI455X (gfx1250) — matrix operand order repaired and checked by reading, not compiled, not run on hardware
//
#include <hip/hip_runtime.h>
#include <hip/hip_bf16.h>

// ---------------------------------------------------------------------------
// Problem constants (reference): B=8, T=4096, H=D=1024, HEADS=1
// ---------------------------------------------------------------------------
#define BB     8
#define TT     4096
#define HH     1024
#define MTOK   (BB * TT)                  // 32768 token rows
#define OUT0   ((size_t)MTOK * HH)        // start of active[:, -1] in d_out
#define OUT1   (OUT0 + (size_t)BB * HH)   // start of ghost[:, -1] in d_out

typedef __attribute__((ext_vector_type(16))) __bf16 v16bf;
typedef __attribute__((ext_vector_type(8)))  float  v8f;

// ---- bf16 <-> f32 helpers (storage is raw unsigned short) ------------------
__device__ __forceinline__ unsigned short f2bf(float f) {
  unsigned int x = __float_as_uint(f);
  unsigned int r = x + 0x7fffu + ((x >> 16) & 1u);   // round-to-nearest-even
  return (unsigned short)(r >> 16);
}
__device__ __forceinline__ float bf2f(unsigned short u) {
  return __uint_as_float(((unsigned int)u) << 16);
}
__device__ __forceinline__ float sigf(float v) {
  return 1.0f / (1.0f + __expf(-v));
}

// ---------------------------------------------------------------------------
// fp32 -> bf16 conversion (weights)
// ---------------------------------------------------------------------------
__global__ void cvt_kernel(const float* __restrict__ src,
                           unsigned short* __restrict__ dst, int n) {
  int i = blockIdx.x * blockDim.x + threadIdx.x;
  if (i < n) dst[i] = f2bf(src[i]);
}

// ---------------------------------------------------------------------------
// LayerNorm: one block per token row, 256 threads x 4 elements
// ---------------------------------------------------------------------------
__global__ __launch_bounds__(256)
void ln_kernel(const float* __restrict__ x, const float* __restrict__ gamma,
               const float* __restrict__ beta, unsigned short* __restrict__ xn) {
  __shared__ float s1[256], s2[256];
  const int row = blockIdx.x, tid = threadIdx.x;
  const float* xr = x + (size_t)row * HH;
  float vals[4], a = 0.f, b = 0.f;
#pragma unroll
  for (int i = 0; i < 4; ++i) {
    float v = xr[tid + 256 * i];
    vals[i] = v; a += v; b += v * v;
  }
  s1[tid] = a; s2[tid] = b;
  __syncthreads();
  for (int off = 128; off; off >>= 1) {
    if (tid < off) { s1[tid] += s1[tid + off]; s2[tid] += s2[tid + off]; }
    __syncthreads();
  }
  const float mu  = s1[0] * (1.0f / HH);
  const float var = s2[0] * (1.0f / HH) - mu * mu;
  const float rs  = rsqrtf(var + 1e-5f);
  unsigned short* xo = xn + (size_t)row * HH;
#pragma unroll
  for (int i = 0; i < 4; ++i) {
    int c = tid + 256 * i;
    xo[c] = f2bf((vals[i] - mu) * rs * gamma[c] + beta[c]);
  }
}

// ---------------------------------------------------------------------------
// u[b,t] = dot(xn_row, Wp_row0) + bp[0]; one wave32 per row
// ---------------------------------------------------------------------------
__global__ __launch_bounds__(256)
void u_kernel(const unsigned short* __restrict__ xn,
              const float* __restrict__ Wp0, const float* __restrict__ bp,
              float* __restrict__ u) {
  const int row  = blockIdx.x * 8 + (threadIdx.x >> 5);
  const int lane = threadIdx.x & 31;
  const unsigned short* xr = xn + (size_t)row * HH;
  float s = 0.f;
#pragma unroll 4
  for (int k = lane; k < HH; k += 32) s += bf2f(xr[k]) * Wp0[k];
#pragma unroll
  for (int m = 16; m; m >>= 1) s += __shfl_xor(s, m, 32);
  if (lane == 0) u[row] = s + bp[0];
}

// ---------------------------------------------------------------------------
// WMMA common bits
// ---------------------------------------------------------------------------
#define LDSS 40   // 32 bf16 + 8 pad (80B row stride, 16B aligned)
union Frag { v16bf v; uint4 q[2]; };

// ---------------------------------------------------------------------------
// Dual-projection bf16 WMMA GEMM: C0 = A*B0^T, C1 = A*B1^T over a shared A
// tile. Block tile 128(M) x 64(N), 8 waves (2x4), wave tile 64x16 per matrix
// = 4 accumulators each. Double-buffered LDS, one barrier per K-step.
// MODE 0 (RV): out0 = f16 sigmoid(C0) [rt],  out1 = f32 C1 + bias [v]
// MODE 1 (GT): out0 = f16 sigmoid(C0)*tanh(C1) [gate*trans]
// ---------------------------------------------------------------------------
template <int MODE>
__global__ __launch_bounds__(256)
void gemm_dual_kernel(const unsigned short* __restrict__ A,
                      const unsigned short* __restrict__ B0,
                      const unsigned short* __restrict__ B1,
                      void* __restrict__ out0, void* __restrict__ out1,
                      const float* __restrict__ bias1, int N, int K) {
  __shared__ unsigned short lsA [2][128][LDSS];
  __shared__ unsigned short lsB0[2][64][LDSS];
  __shared__ unsigned short lsB1[2][64][LDSS];

  const int tid    = threadIdx.x;
  const int lane   = tid & 31;
  const int laneLo = lane & 15;
  const int laneHi = lane >> 4;
  const int wv     = tid >> 5;
  const int waveM  = wv >> 2;          // 0..1 -> 64 rows each
  const int waveN  = wv & 3;           // 0..3 -> 16 cols each
  const int bm     = blockIdx.y * 128;
  const int bn     = blockIdx.x * 64;

  // per-thread staging slots (global -> regs -> LDS)
  const int rA0 = tid >> 2;            // rows 0..63
  const int rA1 = rA0 + 64;            // rows 64..127
  const int cC  = (tid & 3) << 3;      // bf16 col offset 0/8/16/24
  const unsigned short* gA0 = A  + (size_t)(bm + rA0) * K + cC;
  const unsigned short* gA1 = A  + (size_t)(bm + rA1) * K + cC;
  const unsigned short* gB0 = B0 + (size_t)(bn + rA0) * K + cC;  // rA0<64 rows
  const unsigned short* gB1 = B1 + (size_t)(bn + rA0) * K + cC;

  uint4 stA0, stA1, stB0, stB1;
  stA0 = *(const uint4*)(gA0);
  stA1 = *(const uint4*)(gA1);
  stB0 = *(const uint4*)(gB0);
  stB1 = *(const uint4*)(gB1);
  *(uint4*)&lsA [0][rA0][cC] = stA0;
  *(uint4*)&lsA [0][rA1][cC] = stA1;
  *(uint4*)&lsB0[0][rA0][cC] = stB0;
  *(uint4*)&lsB1[0][rA0][cC] = stB1;
  __syncthreads();

  v8f acc0[4] = {}, acc1[4] = {};
  const int KT = K >> 5;
  for (int kt = 0; kt < KT; ++kt) {
    const int cur  = kt & 1;
    const bool more = (kt + 1) < KT;
    if (more) {
      const int kn = (kt + 1) << 5;
      stA0 = *(const uint4*)(gA0 + kn);
      stA1 = *(const uint4*)(gA1 + kn);
      stB0 = *(const uint4*)(gB0 + kn);
      stB1 = *(const uint4*)(gB1 + kn);
      if (kt + 2 < KT) {               // warm near caches for tile k+2
        const int kp = (kt + 2) << 5;
        __builtin_prefetch(gA0 + kp, 0, 3);
        __builtin_prefetch(gA1 + kp, 0, 3);
        __builtin_prefetch(gB0 + kp, 0, 3);
        __builtin_prefetch(gB1 + kp, 0, 3);
      }
    }
    // fragments (A and B: K0-7/K16-23 lanes<16, K8-15/K24-31 lanes>=16)
    Frag a[4], b0, b1;
#pragma unroll
    for (int i = 0; i < 4; ++i) {
      const int row = waveM * 64 + i * 16 + laneLo;
      const int kb  = laneHi << 3;
      a[i].q[0] = *(const uint4*)&lsA[cur][row][kb];
      a[i].q[1] = *(const uint4*)&lsA[cur][row][kb + 16];
    }
    {
      const int row = waveN * 16 + laneLo;
      const int kb  = laneHi << 3;
      b0.q[0] = *(const uint4*)&lsB0[cur][row][kb];
      b0.q[1] = *(const uint4*)&lsB0[cur][row][kb + 16];
      b1.q[0] = *(const uint4*)&lsB1[cur][row][kb];
      b1.q[1] = *(const uint4*)&lsB1[cur][row][kb + 16];
    }
#pragma unroll
    for (int i = 0; i < 4; ++i) {
      acc0[i] = __builtin_amdgcn_wmma_f32_16x16x32_bf16(
          false, a[i].v, false, b0.v, (short)0, acc0[i], false, false);
      acc1[i] = __builtin_amdgcn_wmma_f32_16x16x32_bf16(
          false, a[i].v, false, b1.v, (short)0, acc1[i], false, false);
    }
    if (more) {
      const int nxt = cur ^ 1;
      *(uint4*)&lsA [nxt][rA0][cC] = stA0;
      *(uint4*)&lsA [nxt][rA1][cC] = stA1;
      *(uint4*)&lsB0[nxt][rA0][cC] = stB0;
      *(uint4*)&lsB1[nxt][rA0][cC] = stB1;
    }
    __syncthreads();
  }

  // epilogue: C/D layout -> VGPR r holds M=r (lanes<16) / M=8+r, N=laneLo
  const int col = bn + waveN * 16 + laneLo;
#pragma unroll
  for (int i = 0; i < 4; ++i) {
    const int rowB = bm + waveM * 64 + i * 16 + laneHi * 8;
#pragma unroll
    for (int r = 0; r < 8; ++r) {
      const size_t idx = (size_t)(rowB + r) * N + col;
      const float c0 = acc0[i][r];
      const float c1 = acc1[i][r];
      if (MODE == 0) {
        ((_Float16*)out0)[idx] = (_Float16)sigf(c0);
        ((float*)out1)[idx]    = bias1 ? (c1 + bias1[col]) : c1;
      } else {
        ((_Float16*)out0)[idx] = (_Float16)(sigf(c0) * tanhf(c1));
      }
    }
  }
}

// ---------------------------------------------------------------------------
// Single bf16 WMMA GEMM (output projection): C[M,N] f32 = A[M,K] * W[N,K]^T
// Block tile 128x128, wave tile 64x32 (4x2 accumulators), double-buffered LDS.
// ---------------------------------------------------------------------------
__global__ __launch_bounds__(256)
void gemm_out_kernel(const unsigned short* __restrict__ A,
                     const unsigned short* __restrict__ Bw,
                     float* __restrict__ C, int N, int K) {
  __shared__ unsigned short lsA[2][128][LDSS];
  __shared__ unsigned short lsB[2][128][LDSS];

  const int tid    = threadIdx.x;
  const int lane   = tid & 31;
  const int laneLo = lane & 15;
  const int laneHi = lane >> 4;
  const int wv     = tid >> 5;
  const int waveM  = wv >> 2;
  const int waveN  = wv & 3;
  const int bm     = blockIdx.y * 128;
  const int bn     = blockIdx.x * 128;

  const int rA0 = tid >> 2;
  const int rA1 = rA0 + 64;
  const int cC  = (tid & 3) << 3;
  const unsigned short* gA0 = A  + (size_t)(bm + rA0) * K + cC;
  const unsigned short* gA1 = A  + (size_t)(bm + rA1) * K + cC;
  const unsigned short* gB0 = Bw + (size_t)(bn + rA0) * K + cC;
  const unsigned short* gB1 = Bw + (size_t)(bn + rA1) * K + cC;

  uint4 stA0 = *(const uint4*)(gA0), stA1 = *(const uint4*)(gA1);
  uint4 stB0 = *(const uint4*)(gB0), stB1 = *(const uint4*)(gB1);
  *(uint4*)&lsA[0][rA0][cC] = stA0;
  *(uint4*)&lsA[0][rA1][cC] = stA1;
  *(uint4*)&lsB[0][rA0][cC] = stB0;
  *(uint4*)&lsB[0][rA1][cC] = stB1;
  __syncthreads();

  v8f acc[4][2] = {};
  const int KT = K >> 5;
  for (int kt = 0; kt < KT; ++kt) {
    const int cur   = kt & 1;
    const bool more = (kt + 1) < KT;
    if (more) {
      const int kn = (kt + 1) << 5;
      stA0 = *(const uint4*)(gA0 + kn);
      stA1 = *(const uint4*)(gA1 + kn);
      stB0 = *(const uint4*)(gB0 + kn);
      stB1 = *(const uint4*)(gB1 + kn);
      if (kt + 2 < KT) {
        const int kp = (kt + 2) << 5;
        __builtin_prefetch(gA0 + kp, 0, 3);
        __builtin_prefetch(gB0 + kp, 0, 3);
      }
    }
    Frag a[4], b[2];
#pragma unroll
    for (int i = 0; i < 4; ++i) {
      const int row = waveM * 64 + i * 16 + laneLo;
      const int kb  = laneHi << 3;
      a[i].q[0] = *(const uint4*)&lsA[cur][row][kb];
      a[i].q[1] = *(const uint4*)&lsA[cur][row][kb + 16];
    }
#pragma unroll
    for (int j = 0; j < 2; ++j) {
      const int row = waveN * 32 + j * 16 + laneLo;
      const int kb  = laneHi << 3;
      b[j].q[0] = *(const uint4*)&lsB[cur][row][kb];
      b[j].q[1] = *(const uint4*)&lsB[cur][row][kb + 16];
    }
#pragma unroll
    for (int i = 0; i < 4; ++i)
#pragma unroll
      for (int j = 0; j < 2; ++j)
        acc[i][j] = __builtin_amdgcn_wmma_f32_16x16x32_bf16(
            false, a[i].v, false, b[j].v, (short)0, acc[i][j], false, false);
    if (more) {
      const int nxt = cur ^ 1;
      *(uint4*)&lsA[nxt][rA0][cC] = stA0;
      *(uint4*)&lsA[nxt][rA1][cC] = stA1;
      *(uint4*)&lsB[nxt][rA0][cC] = stB0;
      *(uint4*)&lsB[nxt][rA1][cC] = stB1;
    }
    __syncthreads();
  }

#pragma unroll
  for (int i = 0; i < 4; ++i) {
    const int rowB = bm + waveM * 64 + i * 16 + laneHi * 8;
#pragma unroll
    for (int j = 0; j < 2; ++j) {
      const int col = bn + waveN * 32 + j * 16 + laneLo;
#pragma unroll
      for (int r = 0; r < 8; ++r)
        C[(size_t)(rowB + r) * N + col] = acc[i][j][r];
    }
  }
}

// ---------------------------------------------------------------------------
// Sequential scan over T per (b,d) column: fp32 active cumsum of u*v,
// f16 ghost cumsum (reference accumulates ghost in f16), combine with rt,
// emit bf16 "out" for the final GEMM and the last-state tails.
// ---------------------------------------------------------------------------
__global__ __launch_bounds__(256)
void scan_kernel(const float* __restrict__ u, const float* __restrict__ v,
                 const _Float16* __restrict__ gt, const _Float16* __restrict__ rt,
                 unsigned short* __restrict__ outb, float* __restrict__ dout) {
  const int gid = blockIdx.x * blockDim.x + threadIdx.x;   // 0..8191
  const int b = gid >> 10;
  const int d = gid & 1023;
  size_t idx = (size_t)b * TT * HH + d;
  const int urow = b * TT;
  float    act = 0.f;
  _Float16 gh  = (_Float16)0.0f;
  for (int t = 0; t < TT; ++t) {
    act += u[urow + t] * v[idx];
    gh   = (_Float16)((float)gh + (float)gt[idx]);
    const float comb = act + (float)gh;
    outb[idx] = f2bf((float)rt[idx] * comb);
    idx += HH;
  }
  dout[OUT0 + (size_t)b * HH + d] = act;
  dout[OUT1 + (size_t)b * HH + d] = (float)gh;
}

// ---------------------------------------------------------------------------
// Launch
// ---------------------------------------------------------------------------
extern "C" void kernel_launch(void* const* d_in, const int* in_sizes, int n_in,
                              void* d_out, int out_size, void* d_ws, size_t ws_size,
                              hipStream_t stream) {
  (void)in_sizes; (void)n_in; (void)out_size; (void)ws_size;
  const float* x     = (const float*)d_in[0];
  const float* gamma = (const float*)d_in[1];
  const float* beta  = (const float*)d_in[2];
  const float* Wp    = (const float*)d_in[3];   // (1025, 1024)
  const float* bp    = (const float*)d_in[4];   // (1025,)
  const float* Wg    = (const float*)d_in[5];
  const float* Wt    = (const float*)d_in[6];
  const float* Wr    = (const float*)d_in[7];
  const float* Wo    = (const float*)d_in[8];
  float* out = (float*)d_out;

  // ---- workspace layout ----
  char* ws = (char*)d_ws;
  const size_t SZ_XN = (size_t)MTOK * HH * 2;       // bf16 xn
  const size_t SZ_W  = (size_t)HH * HH * 2;         // bf16 weight
  const size_t SZ_H  = (size_t)MTOK * HH * 2;       // f16 plane
  const size_t SZ_F  = (size_t)MTOK * HH * 4;       // f32 plane
  unsigned short* XN  = (unsigned short*)(ws);      size_t o = SZ_XN;
  unsigned short* WRb = (unsigned short*)(ws + o); o += SZ_W;
  unsigned short* WVb = (unsigned short*)(ws + o); o += SZ_W;
  unsigned short* WGb = (unsigned short*)(ws + o); o += SZ_W;
  unsigned short* WTb = (unsigned short*)(ws + o); o += SZ_W;
  unsigned short* WOb = (unsigned short*)(ws + o); o += SZ_W;
  _Float16* RT   = (_Float16*)(ws + o); o += SZ_H;
  _Float16* GT   = (_Float16*)(ws + o); o += SZ_H;
  float*    Vf   = (float*)(ws + o);    o += SZ_F;
  float*    Uf   = (float*)(ws + o);    o += (size_t)MTOK * 4;
  unsigned short* OUTB = (unsigned short*)(ws + o);

  const int NW = HH * HH;
  // 1) weight conversions (Wp row 0 stays fp32 for the u dot product)
  cvt_kernel<<<NW / 256, 256, 0, stream>>>(Wr,      WRb, NW);
  cvt_kernel<<<NW / 256, 256, 0, stream>>>(Wp + HH, WVb, NW);   // rows 1..1024
  cvt_kernel<<<NW / 256, 256, 0, stream>>>(Wg,      WGb, NW);
  cvt_kernel<<<NW / 256, 256, 0, stream>>>(Wt,      WTb, NW);
  cvt_kernel<<<NW / 256, 256, 0, stream>>>(Wo,      WOb, NW);

  // 2) LayerNorm -> bf16 xn
  ln_kernel<<<MTOK, 256, 0, stream>>>(x, gamma, beta, XN);

  // 3) scalar channel u
  u_kernel<<<MTOK / 8, 256, 0, stream>>>(XN, Wp, bp, Uf);

  // 4) fused front GEMMs: (r|v) and (g|t) share A tiles (M=32768, N=1024, K=1024)
  dim3 dgrid(HH / 64, MTOK / 128);     // (16, 256)
  gemm_dual_kernel<0><<<dgrid, 256, 0, stream>>>(XN, WRb, WVb, (void*)RT, (void*)Vf, bp + 1, HH, HH);
  gemm_dual_kernel<1><<<dgrid, 256, 0, stream>>>(XN, WGb, WTb, (void*)GT, nullptr,  nullptr, HH, HH);

  // 5) recurrent scans + combine (+ last-state tails into d_out)
  scan_kernel<<<(BB * HH) / 256, 256, 0, stream>>>(Uf, Vf, GT, RT, OUTB, out);

  // 6) output projection straight into d_out
  dim3 ogrid(HH / 128, MTOK / 128);    // (8, 256)
  gemm_out_kernel<<<ogrid, 256, 0, stream>>>(OUTB, WOb, out, HH, HH);
}
